// SPF_LUT_net_10101763080280
// MI455X (gfx1250) — matrix operand order repaired and checked by reading, not compiled, not run on hardware
//
#include <hip/hip_runtime.h>
#include <math.h>

// ---------------------------------------------------------------------------
// CDNA5 (gfx1250, wave32) implementation of SPF-LUT forward.
// All GEMM work runs through v_wmma_f32_16x16x32_f16 (M=16 pixels, N=16 feats,
// K=32), f32 accumulate, f16 operands staged in LDS. Unit weight blocks are
// staged to LDS with the Tensor Data Mover (6-arg builtin on this toolchain).
// ---------------------------------------------------------------------------

typedef __attribute__((ext_vector_type(16))) _Float16 v16h;
typedef __attribute__((ext_vector_type(8)))  float    v8f;
typedef __attribute__((ext_vector_type(4)))  unsigned int u32x4;
typedef __attribute__((ext_vector_type(8)))  int      i32x8;
typedef __attribute__((ext_vector_type(4)))  int      i32x4;

struct alignas(16) F4 { unsigned int x, y, z, w; };
union Frag { F4 q[2]; v16h h; _Float16 e[16]; unsigned int u[8]; };

#define UNIT_BYTES  28672   // packed stride per unit in global (padded)
#define UNIT_LIVE   28352   // 13824 halfs (27648 B) + 176 floats (704 B)
#define UNIT_WORDS  7088    // UNIT_LIVE / 4
#define W_HALFS     13824   // W1(32x32) W2(32x32) W3(32x64) W4(32x96) W5(32x128) W6(16x160)
#define B_FLOATS    176     // b1..b5 (32 each) + b6 (16)
#define H_STRIDE    160     // h row stride in halfs (h1..h5 concatenated)
#define NUNITS      25      // 4 stages*3 + up 12 + cc 1

#if defined(__gfx1250__) && __has_builtin(__builtin_amdgcn_tensor_load_to_lds) && \
    __has_builtin(__builtin_amdgcn_s_wait_tensorcnt)
#define USE_TDM 1
#else
#define USE_TDM 0
#endif

__device__ __forceinline__ v8f wmma16(v16h a, v16h b, v8f c) {
  return __builtin_amdgcn_wmma_f32_16x16x32_f16(false, a, false, b, (short)0, c,
                                                false, false);
}

// A fragment (16x32 f16): lane<16 holds K {kbase+0..7, kbase+16..23} of row m=lane&15,
// lane>=16 holds K {kbase+8..15, kbase+24..31}.
__device__ __forceinline__ v16h ldsA(const _Float16* hrow, int kbase, int lane) {
  int off = kbase + ((lane >> 4) << 3);
  Frag f;
  f.q[0] = *(const F4*)(hrow + off);
  f.q[1] = *(const F4*)(hrow + off + 16);
  return f.h;
}

// B fragment (32x16 f16): lane&15 selects column (output feature row of W),
// lane<16 holds K {kbase+0..7, kbase+16..23}, lane>=16 holds K {kbase+8..15, kbase+24..31}.
__device__ __forceinline__ v16h ldsB(const _Float16* W, int K, int row, int kbase,
                                     int lane) {
  const _Float16* p = W + row * K + kbase + ((lane >> 4) << 3);
  Frag f;
  f.q[0] = *(const F4*)(p);
  f.q[1] = *(const F4*)(p + 16);
  return f.h;
}

// D layout: lane&15 = N column, rows M = ((lane>>4)<<3) + v.
__device__ __forceinline__ void store_relu(_Float16* hwave, int dstcol, v8f acc,
                                           float bias, int mrow, int n) {
#pragma unroll
  for (int v = 0; v < 8; ++v) {
    float x = acc[v] + bias;
    hwave[(mrow + v) * H_STRIDE + dstcol + n] = (_Float16)(x > 0.f ? x : 0.f);
  }
}

// Conv A fragment: K0..3 = x taps, K4..7 = prev taps, K16 = 1.0 (bias lane),
// rest 0. Lanes>=16 all zero.
__device__ __forceinline__ v16h build_conv_frag(const float* xim, const float* pim,
                                                int i, int j, int r, int lane) {
  Frag f;
#pragma unroll
  for (int w = 0; w < 8; ++w) f.u[w] = 0u;
  if (lane < 16) {
    static const int DI[4][4] = {{0,0,1,1},{0,1,0,1},{0,0,-1,-1},{0,-1,0,-1}};
    static const int DJ[4][4] = {{0,1,0,1},{0,0,-1,-1},{0,-1,0,-1},{0,0,1,1}};
#pragma unroll
    for (int t = 0; t < 4; ++t) {
      int ti = i + DI[r][t]; ti = ti < 0 ? 0 : (ti > 127 ? 127 : ti);
      int tj = j + DJ[r][t]; tj = tj < 0 ? 0 : (tj > 127 ? 127 : tj);
      f.e[t] = (_Float16)xim[ti * 128 + tj];
      f.e[4 + t] = pim ? (_Float16)pim[ti * 128 + tj] : (_Float16)0.f;
    }
    f.e[8] = (_Float16)1.0f;  // K=16: constant-1 feature carrying b1 via WMMA
  }
  return f.h;
}

// Stage one packed unit (weights f16 + biases f32, 28352 B contiguous) to LDS.
// Preferred path: single Tensor Data Mover 1-D tile issued by wave 0.
__device__ __forceinline__ void stage_unit(const unsigned char* wpack, int p,
                                           unsigned char* sUnit, int tid) {
#if USE_TDM
  if (tid < 32) {
    unsigned long long ga = (unsigned long long)(wpack + (size_t)p * UNIT_BYTES);
    unsigned int laddr = (unsigned int)(unsigned long long)(uintptr_t)sUnit;
    u32x4 g0;
    g0[0] = 1u;                                       // count=1, user mode
    g0[1] = laddr;                                    // lds_addr
    g0[2] = (unsigned int)ga;                         // global_addr[31:0]
    g0[3] = (unsigned int)((ga >> 32) & 0x01FFFFFFull) | (2u << 30); // +type=2
    i32x8 g1;
    g1[0] = (int)(2u << 16);                          // data_size=4B, mask=0
    g1[1] = (int)((UNIT_WORDS & 0xFFFF) << 16);       // tensor_dim0[15:0]
    g1[2] = (int)((UNIT_WORDS >> 16) & 0xFFFF);       // tensor_dim0[31:16], dim1=0
    g1[3] = (int)((UNIT_WORDS & 0xFFFF) << 16);       // tile_dim0 = UNIT_WORDS
    g1[4] = 0;                                        // tile_dim1/2 unused
    g1[5] = (int)UNIT_WORDS;                          // tensor_dim0_stride lo
    g1[6] = 0;
    g1[7] = 0;
    i32x4 z4; z4[0] = 0; z4[1] = 0; z4[2] = 0; z4[3] = 0;
    i32x8 z8;
#pragma unroll
    for (int w = 0; w < 8; ++w) z8[w] = 0;
    // 6-arg toolchain variant: (g0, g1, g2, g3, g4, cpol)
    __builtin_amdgcn_tensor_load_to_lds(g0, g1, z4, z4, z8, 0);
    __builtin_amdgcn_s_wait_tensorcnt((short)0);
  }
#else
  const unsigned int* src = (const unsigned int*)(wpack + (size_t)p * UNIT_BYTES);
  unsigned int* dst = (unsigned int*)sUnit;
  for (int e = tid; e < UNIT_WORDS; e += 128) dst[e] = src[e];
#endif
}

// Full unit: conv(WMMA, bias folded) -> dense chain (WMMA, A frags hoisted
// across N-tiles) -> W6 (WMMA) -> round(tanh*127).
__device__ __forceinline__ v8f unit_forward(v16h aConv, const _Float16* sW,
                                            const float* sB, _Float16* hwave,
                                            const _Float16* myHrow, int lane) {
  int n = lane & 15;
  int mrow = (lane >> 4) << 3;
  // conv layer -> h1 (cols 0..31); bias carried by constant-1 K lane
#pragma unroll
  for (int nt = 0; nt < 2; ++nt) {
    v8f acc = {};
    v16h b = ldsB(sW, 32, nt * 16 + n, 0, lane);
    acc = wmma16(aConv, b, acc);
    store_relu(hwave, nt * 16, acc, 0.f, mrow, n);
  }
  constexpr int Ks[4]   = {32, 64, 96, 128};
  constexpr int Woff[4] = {1024, 2048, 4096, 7168};
#pragma unroll
  for (int l = 0; l < 4; ++l) {
    const int K = Ks[l];          // input width == dest column base
    v16h afr[4];
#pragma unroll
    for (int kc = 0; kc < K; kc += 32) afr[kc >> 5] = ldsA(myHrow, kc, lane);
#pragma unroll
    for (int nt = 0; nt < 2; ++nt) {
      v8f acc = {};
#pragma unroll
      for (int kc = 0; kc < K; kc += 32) {
        v16h b = ldsB(sW + Woff[l], K, nt * 16 + n, kc, lane);
        acc = wmma16(afr[kc >> 5], b, acc);
      }
      store_relu(hwave, K + nt * 16, acc, sB[K + nt * 16 + n], mrow, n);
    }
  }
  // W6: 160 -> 16 (rows >= out_c are zero-padded)
  v8f acc = {};
#pragma unroll
  for (int kc = 0; kc < 160; kc += 32) {
    v16h a = ldsA(myHrow, kc, lane);
    v16h b = ldsB(sW + 11264, 160, n, kc, lane);
    acc = wmma16(a, b, acc);
  }
  float b6 = sB[160 + n];
  v8f y;
#pragma unroll
  for (int v = 0; v < 8; ++v) y[v] = rintf(tanhf(acc[v] + b6) * 127.f);
  return y;
}

// ---------------------------------------------------------------------------
// Parameter packing: fp32 leaves -> f16 WMMA-friendly layout, masks folded in,
// b1 folded into W1 column k=16 (matches constant-1 conv feature).
// ---------------------------------------------------------------------------
struct PtrTable { const float* p[326]; };

__global__ void pack_params_kernel(PtrTable T, unsigned char* wpack, int n_in) {
  int p = blockIdx.x;  // packed unit id 0..24
  int sortedMode[3] = {1, 0, 2};  // compute order (s,d,y) -> sorted (d,s,y)
  int base, outc, mode;
  if (p < 12)       { int st = p / 3; mode = p % 3; base = 1 + st * 39 + sortedMode[mode] * 13; outc = (st == 3) ? 1 : 2; }
  else if (p < 24)  { int u = p - 12; int c = u / 3; mode = u % 3; base = 169 + (c * 3 + sortedMode[mode]) * 13; outc = 16; }
  else              { base = 157; mode = -1; outc = 4; }

  auto G = [&](int off) -> const float* {
    int i = base + off;
    if (i >= n_in) i = 0;
    return T.p[i];
  };
  const float *b1 = G(0), *b2 = G(1), *b3 = G(2), *b4 = G(3), *b5 = G(4), *b6 = G(5);
  const float *w1, *w1p, *w2, *w3, *w4, *w5, *w6;
  if (p == 24) { w1 = G(6); w1p = nullptr; w2 = G(7); w3 = G(8); w4 = G(9); w5 = G(10); w6 = G(11); }
  else         { w1 = G(6); w1p = G(7);    w2 = G(8); w3 = G(9); w4 = G(10); w5 = G(11); w6 = G(12); }

  // 2x2 mask per mode; K=2 makes 's' and 'd' both all-ones; 'y' kills tap (0,1).
  float mask[4] = {1.f, 1.f, 1.f, 1.f};
  if (mode == 2) mask[1] = 0.f;

  _Float16* W = (_Float16*)(wpack + (size_t)p * UNIT_BYTES);
  float*    B = (float*)(wpack + (size_t)p * UNIT_BYTES + W_HALFS * 2);

  for (int e = threadIdx.x; e < W_HALFS; e += blockDim.x) {
    float val = 0.f;
    if (e < 1024) {            // W1 [32][32]: k0..3 = w1*mask, k4..7 = w1p*mask, k16 = b1
      int nn = e >> 5, k = e & 31;
      if (k < 4)                 val = w1[nn * 4 + k] * mask[k];
      else if (k < 8 && w1p)     val = w1p[nn * 4 + (k - 4)] * mask[k - 4];
      else if (k == 16)          val = b1[nn];
    } else if (e < 2048)  { int t = e - 1024; val = w2[(t >> 5) * 32 + (t & 31)]; }
    else if (e < 4096)    { int t = e - 2048; val = w3[(t / 64) * 64 + (t % 64)]; }
    else if (e < 7168)    { int t = e - 4096; val = w4[(t / 96) * 96 + (t % 96)]; }
    else if (e < 11264)   { int t = e - 7168; val = w5[(t / 128) * 128 + (t % 128)]; }
    else { int t = e - 11264; int nn = t / 160, k = t % 160; if (nn < outc) val = w6[nn * 160 + k]; }
    W[e] = (_Float16)val;
  }
  for (int e = threadIdx.x; e < B_FLOATS; e += blockDim.x) {
    float val = 0.f;
    if (e < 32)        val = b1[e];
    else if (e < 64)   val = b2[e - 32];
    else if (e < 96)   val = b3[e - 64];
    else if (e < 128)  val = b4[e - 96];
    else if (e < 160)  val = b5[e - 128];
    else { int nn = e - 160; if (nn < outc) val = b6[nn]; }
    B[e] = val;
  }
}

// ---------------------------------------------------------------------------
// cb1..cb4: in_c=1, scale=1. 64 pixels per block (4 waves x 16-pixel M-tile).
// Accumulate 12 (mode x rot) contributions in registers, then quant.
// ---------------------------------------------------------------------------
__global__ __launch_bounds__(128) void cb_stage_kernel(
    const unsigned char* __restrict__ wpack, int punitBase,
    const float* __restrict__ xin, const float* __restrict__ prevIn,
    float* __restrict__ refineOut, int refineImgStride,
    float* __restrict__ nextOut, int outc) {
  __shared__ alignas(16) unsigned char sUnit[UNIT_LIVE];
  __shared__ alignas(16) _Float16 sH[4][16][H_STRIDE];
  _Float16* sW = (_Float16*)sUnit;
  float*    sB = (float*)(sUnit + W_HALFS * 2);

  int img = blockIdx.y;
  int tid = threadIdx.x, lane = tid & 31, wv = tid >> 5;
  int n = lane & 15;
  int pix0 = blockIdx.x * 64 + wv * 16;
  int myPix = pix0 + n;
  int i = myPix >> 7, j = myPix & 127;
  const float* xim = xin + img * 16384;
  const float* pim = prevIn ? prevIn + img * 16384 : nullptr;
  _Float16* hwave = &sH[wv][0][0];
  const _Float16* myHrow = &sH[wv][n][0];

  v8f total = {};
  for (int mode = 0; mode < 3; ++mode) {
    __syncthreads();
    stage_unit(wpack, punitBase + mode, sUnit, tid);
    __syncthreads();
#pragma unroll
    for (int r = 0; r < 4; ++r) {
      v16h a = build_conv_frag(xim, pim, i, j, r, lane);
      v8f y = unit_forward(a, sW, sB, hwave, myHrow, lane);
      total = total + y;
    }
  }
  // quant(z) = round(clip(z/12 + 127, 0, 255)) / 255
  if (n < outc) {
    int mrow = (lane >> 4) << 3;
#pragma unroll
    for (int v = 0; v < 8; ++v) {
      float q = total[v] * (1.f / 12.f) + 127.f;
      q = rintf(fminf(fmaxf(q, 0.f), 255.f)) * (1.f / 255.f);
      int pix = pix0 + mrow + v;
      if (n == 0)             refineOut[(size_t)img * refineImgStride + pix] = q;
      else if (nextOut)       nextOut[(size_t)img * 16384 + pix] = q;
    }
  }
}

// ---------------------------------------------------------------------------
// cc: pointwise 4->32 then dense chain, 4 outputs re-quantized to [0,1].
// ---------------------------------------------------------------------------
__global__ __launch_bounds__(128) void cc_kernel(
    const unsigned char* __restrict__ wpack,
    const float* __restrict__ refine, float* __restrict__ ccout) {
  __shared__ alignas(16) unsigned char sUnit[UNIT_LIVE];
  __shared__ alignas(16) _Float16 sH[4][16][H_STRIDE];
  _Float16* sW = (_Float16*)sUnit;
  float*    sB = (float*)(sUnit + W_HALFS * 2);

  int img = blockIdx.y;
  int tid = threadIdx.x, lane = tid & 31, wv = tid >> 5;
  int n = lane & 15;
  int pix0 = blockIdx.x * 64 + wv * 16;
  int myPix = pix0 + n;

  __syncthreads();
  stage_unit(wpack, 24, sUnit, tid);
  __syncthreads();

  Frag f;
#pragma unroll
  for (int w = 0; w < 8; ++w) f.u[w] = 0u;
  if (lane < 16) {
#pragma unroll
    for (int t = 0; t < 4; ++t)
      f.e[t] = (_Float16)refine[(size_t)img * 65536 + t * 16384 + myPix];
    f.e[8] = (_Float16)1.0f;  // bias feature
  }
  v8f y = unit_forward(f.h, sW, sB, &sH[wv][0][0], &sH[wv][n][0], lane);

  if (n < 4) {
    int mrow = (lane >> 4) << 3;
#pragma unroll
    for (int v = 0; v < 8; ++v) {
      float q = rintf(fminf(fmaxf(y[v] + 127.f, 0.f), 255.f)) * (1.f / 255.f);
      ccout[(size_t)img * 65536 + n * 16384 + pix0 + mrow + v] = q;
    }
  }
}

// ---------------------------------------------------------------------------
// up: in_c=4, scale=4, out_c=16. Accumulate 12 (c x mode) per rotation in
// registers; scatter via pixel-shuffle + inverse rotation to per-r buffers
// (bijective per r -> single writer, plain stores).
// ---------------------------------------------------------------------------
__global__ __launch_bounds__(128) void up_kernel(
    const unsigned char* __restrict__ wpack, const float* __restrict__ ccout,
    const float* __restrict__ prev0, const float* __restrict__ prev1,
    const float* __restrict__ prev2, const float* __restrict__ prev3,
    float* __restrict__ upbuf) {
  __shared__ alignas(16) unsigned char sUnit[UNIT_LIVE];
  __shared__ alignas(16) _Float16 sH[4][16][H_STRIDE];
  _Float16* sW = (_Float16*)sUnit;
  float*    sB = (float*)(sUnit + W_HALFS * 2);

  int img = blockIdx.y;
  int tid = threadIdx.x, lane = tid & 31, wv = tid >> 5;
  int n = lane & 15;
  int pix0 = blockIdx.x * 64 + wv * 16;
  int myPix = pix0 + n;
  int i = myPix >> 7, j = myPix & 127;
  _Float16* hwave = &sH[wv][0][0];
  const _Float16* myHrow = &sH[wv][n][0];
  const float* prevs[4] = {prev0, prev1, prev2, prev3};

  v8f acc0 = {}, acc1 = {}, acc2 = {}, acc3 = {};
  for (int c = 0; c < 4; ++c) {
    const float* xim = ccout + (size_t)img * 65536 + c * 16384;
    const float* pim = prevs[c] + (size_t)img * 16384;
    for (int mode = 0; mode < 3; ++mode) {
      __syncthreads();
      stage_unit(wpack, 12 + c * 3 + mode, sUnit, tid);
      __syncthreads();
      {
        v16h a = build_conv_frag(xim, pim, i, j, 0, lane);
        acc0 = acc0 + unit_forward(a, sW, sB, hwave, myHrow, lane);
      }
      {
        v16h a = build_conv_frag(xim, pim, i, j, 1, lane);
        acc1 = acc1 + unit_forward(a, sW, sB, hwave, myHrow, lane);
      }
      {
        v16h a = build_conv_frag(xim, pim, i, j, 2, lane);
        acc2 = acc2 + unit_forward(a, sW, sB, hwave, myHrow, lane);
      }
      {
        v16h a = build_conv_frag(xim, pim, i, j, 3, lane);
        acc3 = acc3 + unit_forward(a, sW, sB, hwave, myHrow, lane);
      }
    }
  }

  int q = n;              // pixel-shuffle channel: q = ry*4 + rx
  int ry = q >> 2, rx = q & 3;
  int mrow = (lane >> 4) << 3;
#pragma unroll
  for (int r = 0; r < 4; ++r) {
    v8f acc = (r == 0) ? acc0 : (r == 1) ? acc1 : (r == 2) ? acc2 : acc3;
    int s = (4 - r) & 3;  // inverse rotation on 512 grid
#pragma unroll
    for (int v = 0; v < 8; ++v) {
      int pix = pix0 + mrow + v;
      int pi = pix >> 7, pj = pix & 127;
      int u, vv;
      if (r == 0)      { u = pi;        vv = pj; }
      else if (r == 1) { u = 127 - pj;  vv = pi; }
      else if (r == 2) { u = 127 - pi;  vv = 127 - pj; }
      else             { u = pj;        vv = 127 - pi; }
      int U = 4 * u + ry, V = 4 * vv + rx;
      int I, J;
      if (s == 0)      { I = U;        J = V; }
      else if (s == 1) { I = 511 - V;  J = U; }
      else if (s == 2) { I = 511 - U;  J = 511 - V; }
      else             { I = V;        J = 511 - U; }
      upbuf[(size_t)r * 786432 + (size_t)img * 262144 + I * 512 + J] = acc[v];
    }
  }
}

__global__ void combine_kernel(const float* __restrict__ upbuf,
                               float* __restrict__ out) {
  int idx = blockIdx.x * 256 + threadIdx.x;
  if (idx >= 786432) return;
  float s = upbuf[idx] + upbuf[786432 + idx] + upbuf[2 * 786432 + idx] +
            upbuf[3 * 786432 + idx];
  out[idx] = rintf(s * (1.f / 12.f)) * (1.f / 255.f);
}

// ---------------------------------------------------------------------------
// Host orchestration (all on `stream`, graph-capture safe: launches only).
// ---------------------------------------------------------------------------
extern "C" void kernel_launch(void* const* d_in, const int* in_sizes, int n_in,
                              void* d_out, int out_size, void* d_ws, size_t ws_size,
                              hipStream_t stream) {
  (void)in_sizes; (void)out_size; (void)ws_size;
  const float* x = (const float*)d_in[0];   // (1,3,128,128) -> 3 images of 16384
  unsigned char* ws = (unsigned char*)d_ws;

  unsigned char* wpack = ws;                                  // 25 * 28672
  float* refine = (float*)(ws + 720896);                      // [3][4][16384]
  float* x2     = (float*)(ws + 1507328);                     // [3][16384]
  float* x3     = (float*)(ws + 1703936);
  float* x4     = (float*)(ws + 1900544);
  float* ccout  = (float*)(ws + 2097152);                     // [3][4][16384]
  float* upbuf  = (float*)(ws + 2883584);                     // [4][3][512*512]

  PtrTable T;
  for (int t = 0; t < 326; ++t)
    T.p[t] = (const float*)d_in[(t < n_in) ? t : 0];

  pack_params_kernel<<<NUNITS, 256, 0, stream>>>(T, wpack, n_in);

  dim3 grid(256, 3);  // 256 blocks x 64 pixels = 16384 per image, 3 images
  cb_stage_kernel<<<grid, 128, 0, stream>>>(wpack, 0, x,  nullptr, refine + 0 * 16384, 65536, x2, 2);
  cb_stage_kernel<<<grid, 128, 0, stream>>>(wpack, 3, x2, x,       refine + 1 * 16384, 65536, x3, 2);
  cb_stage_kernel<<<grid, 128, 0, stream>>>(wpack, 6, x3, x2,      refine + 2 * 16384, 65536, x4, 2);
  cb_stage_kernel<<<grid, 128, 0, stream>>>(wpack, 9, x4, x3,      refine + 3 * 16384, 65536, nullptr, 1);
  cc_kernel<<<grid, 128, 0, stream>>>(wpack, refine, ccout);
  up_kernel<<<grid, 128, 0, stream>>>(wpack, ccout, x, x2, x3, x4, upbuf);
  combine_kernel<<<3072, 256, 0, stream>>>(upbuf, (float*)d_out);
}
